// vvbp_tensor_net_48627619725593
// MI455X (gfx1250) — matrix operand order repaired and checked by reading, not compiled, not run on hardware
//
#include <hip/hip_runtime.h>
#include <hip/hip_bf16.h>
#include <math.h>

typedef __attribute__((ext_vector_type(16))) _Float16 v16h;
typedef __attribute__((ext_vector_type(8)))  _Float16 v8h;
typedef __attribute__((ext_vector_type(8)))  float    v8f;

// Problem constants (match reference)
#define NB   2
#define NV   256
#define ND   512
#define IMG  256
#define PIX  (IMG * IMG)   // 65536
#define CH   96

// ---------------------------------------------------------------------------
// Repack 96x96x5x5 f32 OIHW weights into WMMA A-fragment order (f16):
// layout [tap(25)][icb(3)][octile(6)][lane(32)][elem(16)], using contiguous k order:
// k = 16*(lane/16) + elem (same order as the B loads in conv96_wmma).
// ---------------------------------------------------------------------------
__global__ __launch_bounds__(256) void repack_w96(const float* __restrict__ w,
                                                  _Float16* __restrict__ dst) {
  int tid = blockIdx.x * blockDim.x + threadIdx.x;
  if (tid >= 25 * 3 * 6 * 32 * 16) return;
  int e    = tid & 15;
  int lane = (tid >> 4) & 31;
  int t2   = tid >> 9;       // (tap*3+icb)*6 + oct
  int oct  = t2 % 6;
  int t3   = t2 / 6;         // tap*3 + icb
  int icb  = t3 % 3;
  int tap  = t3 / 3;
  int j = e >> 1, hh = e & 1;
  int koff = (lane & 16) ? 16 : 0;
  int K = e + koff;
  int oc = oct * 16 + (lane & 15);
  int ic = icb * 32 + K;
  int ky = tap / 5, kx = tap % 5;
  dst[tid] = (_Float16)w[((oc * CH + ic) * 5 + ky) * 5 + kx];
}

// ---------------------------------------------------------------------------
// Sinogram weighting + Ram-Lak ramp filter (1023-tap correlation along D).
// ---------------------------------------------------------------------------
__global__ __launch_bounds__(256) void sino_filter(const float* __restrict__ p,
                                                   const float* __restrict__ pw,
                                                   const float* __restrict__ ramp,
                                                   float* __restrict__ filtered) {
  __shared__ float row[ND];
  __shared__ float rf[1024];
  int v = blockIdx.x, b = blockIdx.y, t = threadIdx.x;
  const float* pr  = p  + ((size_t)b * NV + v) * ND;
  const float* pwr = pw + (size_t)v * ND;
  row[t]       = pr[t]       * pwr[t];
  row[t + 256] = pr[t + 256] * pwr[t + 256];
  for (int i = t; i < 2 * ND - 1; i += 256) rf[i] = ramp[i];
  __syncthreads();
  for (int d = t; d < ND; d += 256) {
    float acc = 0.f;
    for (int j = 0; j < 2 * ND - 1; ++j) {
      int idx = d + j - (ND - 1);
      if (idx >= 0 && idx < ND) acc = fmaf(row[idx], rf[j], acc);
    }
    filtered[((size_t)b * NV + v) * ND + d] = acc;
  }
}

// ---------------------------------------------------------------------------
// Per-view backprojection: [B,V,H,W], linear interpolation on the detector.
// ---------------------------------------------------------------------------
__global__ __launch_bounds__(256) void backproject(const float* __restrict__ filtered,
                                                   const float* __restrict__ views,
                                                   float* __restrict__ xout) {
  int pix = blockIdx.x * 256 + threadIdx.x;
  int v = blockIdx.y, b = blockIdx.z;
  int h = pix >> 8, w = pix & 255;
  float sn, cs;
  __sincosf(views[v], &sn, &cs);
  float X = (float)w - 127.5f, Y = (float)h - 127.5f;
  float u  = X * cs + Y * sn + (float)(ND - 1) * 0.5f;
  float uf = floorf(u);
  int u0 = (int)uf;
  float frac = u - uf;
  bool valid = (u0 >= 0) && (u0 <= ND - 2);
  int uc = min(max(u0, 0), ND - 2);
  const float* fr = filtered + ((size_t)b * NV + v) * ND;
  float g0 = fr[uc], g1 = fr[uc + 1];
  float g = valid ? (g0 + (g1 - g0) * frac) : 0.f;
  xout[((size_t)(b * NV + v)) * PIX + pix] = g * 0.01227184630f;  // pi/256
}

// ---------------------------------------------------------------------------
// In-place bitonic sort across the view dimension (256 values / pixel).
// 32 pixels per workgroup, LDS stride 257 floats to avoid bank conflicts.
// ---------------------------------------------------------------------------
__global__ __launch_bounds__(256) void sort_views(float* __restrict__ x) {
  __shared__ float s[32 * 257];
  int t = threadIdx.x;
  int pix0 = blockIdx.x * 32;
  int b = blockIdx.y;
  size_t base = (size_t)b * NV * PIX + pix0;
  for (int m = 0; m < 32; ++m) {
    int e = t + m * 256;
    int v = e >> 5, pq = e & 31;
    s[pq * 257 + v] = x[base + (size_t)v * PIX + pq];
  }
  __syncthreads();
  for (int k = 2; k <= 256; k <<= 1) {
    for (int j = k >> 1; j > 0; j >>= 1) {
      for (int m = 0; m < 32; ++m) {
        int e = t + m * 256;
        int pq = e & 31;
        int i = e >> 5;
        int ixj = i ^ j;
        if (ixj > i) {
          float a = s[pq * 257 + i], c = s[pq * 257 + ixj];
          bool up = (i & k) == 0;
          if ((a > c) == up) { s[pq * 257 + i] = c; s[pq * 257 + ixj] = a; }
        }
      }
      __syncthreads();
    }
  }
  for (int m = 0; m < 32; ++m) {
    int e = t + m * 256;
    int v = e >> 5, pq = e & 31;
    x[base + (size_t)v * PIX + pq] = s[pq * 257 + v];
  }
}

// ---------------------------------------------------------------------------
// 256-channel -> 1 channel 3x3 conv (view-mixing). Weights cached in LDS.
// ---------------------------------------------------------------------------
__global__ __launch_bounds__(256) void conv_views(const float* __restrict__ xs,
                                                  const float* __restrict__ w,
                                                  const float* __restrict__ bias,
                                                  float* __restrict__ xin) {
  __shared__ float ws[NV * 9];
  int t = threadIdx.x;
  for (int i = t; i < NV * 9; i += 256) ws[i] = w[i];
  __syncthreads();
  int pix = blockIdx.x * 256 + t;
  int b = blockIdx.y;
  int y = pix >> 8, x = pix & 255;
  float acc = bias[0];
  const float* xb = xs + (size_t)b * NV * PIX;
  for (int v = 0; v < NV; ++v) {
    const float* xv = xb + (size_t)v * PIX;
#pragma unroll
    for (int ky = 0; ky < 3; ++ky) {
      int gy = y + ky - 1;
      if (gy < 0 || gy > IMG - 1) continue;
#pragma unroll
      for (int kx = 0; kx < 3; ++kx) {
        int gx = x + kx - 1;
        if (gx < 0 || gx > IMG - 1) continue;
        acc = fmaf(xv[gy * IMG + gx], ws[v * 9 + ky * 3 + kx], acc);
      }
    }
  }
  xin[(size_t)b * PIX + pix] = acc;
}

// ---------------------------------------------------------------------------
// RED-CNN head: 1 -> 96 channels, 5x5, bias + ReLU.
// One thread per pixel, all 96 oc; NHWC f16 output (96 contiguous halves).
// ---------------------------------------------------------------------------
__global__ __launch_bounds__(256) void conv_head(const float* __restrict__ xin,
                                                 const float* __restrict__ w0,
                                                 const float* __restrict__ bias,
                                                 _Float16* __restrict__ act) {
  __shared__ float ws[CH * 25];
  __shared__ float bs[CH];
  int t = threadIdx.x;
  for (int i = t; i < CH * 25; i += 256) ws[i] = w0[i];
  if (t < CH) bs[t] = bias[t];
  __syncthreads();
  int pix = blockIdx.x * 256 + t;
  int b = blockIdx.y;
  int y = pix >> 8, x = pix & 255;
  const float* xb = xin + (size_t)b * PIX;
  float patch[25];
#pragma unroll
  for (int ky = 0; ky < 5; ++ky)
#pragma unroll
    for (int kx = 0; kx < 5; ++kx) {
      int gy = y + ky - 2, gx = x + kx - 2;
      bool v = (gy >= 0) && (gy < IMG) && (gx >= 0) && (gx < IMG);
      patch[ky * 5 + kx] = v ? xb[gy * IMG + gx] : 0.f;
    }
  _Float16* ap = act + ((size_t)b * PIX + pix) * CH;
  for (int oc = 0; oc < CH; ++oc) {
    float acc = bs[oc];
#pragma unroll
    for (int k = 0; k < 25; ++k) acc = fmaf(patch[k], ws[oc * 25 + k], acc);
    ap[oc] = (_Float16)fmaxf(acc, 0.f);
  }
}

// ---------------------------------------------------------------------------
// Main WMMA kernel: 96 -> 96 channels, 5x5, SAME, implicit GEMM, NHWC f16.
// B fragments load DIRECTLY from global: k order used here (same as A pack) = 16
// consecutive K (input channels) per lane for one pixel -> contiguous 32B in
// NHWC. No LDS, no barriers. Each wave: 2 pixel groups x 3 oc tiles ->
// 6 accumulators, 6 v_wmma_f32_16x16x32_f16 per k-step (75 k-steps).
// Out-of-range rows skipped uniformly; out-of-range columns zeroed per lane
// via cndmask (EXEC stays all-1s at every WMMA).
// ---------------------------------------------------------------------------
__global__ __launch_bounds__(256) void conv96_wmma(const _Float16* __restrict__ in,
                                                   const _Float16* __restrict__ wpack,
                                                   const float* __restrict__ bias,
                                                   const _Float16* __restrict__ skip,
                                                   _Float16* __restrict__ out) {
  const int tid   = threadIdx.x;
  const int lane  = tid & 31;
  const int wave  = tid >> 5;
  const int pg2   = wave & 3;   // pixel-pair group: pixels [pg2*32, pg2*32+32)
  const int ocset = wave >> 2;  // 0: oc tiles 0..2, 1: oc tiles 3..5

  const int x0 = blockIdx.x * 128;
  const int y  = blockIdx.y;
  const size_t img = (size_t)blockIdx.z * CH * PIX;

  const int nl = lane & 15;         // N within 16-pixel tile
  const int kh = (lane >> 4) * 16;  // K-half offset within 32 (B layout)
  const int pxA = x0 + pg2 * 32 + nl;
  const int pxB = pxA + 16;

  v8f acc[6] = {};  // [g*3 + i]: g = pixel group, i = oc tile
  const v16h vzero = {};

  for (int ky = -2; ky <= 2; ++ky) {
    const int gy = y + ky;
    if (gy < 0 || gy >= IMG) continue;  // block-uniform row skip
    const _Float16* rowp = in + img + (size_t)gy * (IMG * CH) + kh;
    for (int kx = -2; kx <= 2; ++kx) {
      const int tap = (ky + 2) * 5 + (kx + 2);
      const int gxA = pxA + kx, gxB = pxB + kx;
      const bool vA = (gxA >= 0) && (gxA < IMG);
      const bool vB = (gxB >= 0) && (gxB < IMG);
      const int cA = min(max(gxA, 0), IMG - 1) * CH;
      const int cB = min(max(gxB, 0), IMG - 1) * CH;
      const _Float16* wtap =
          wpack + (size_t)(tap * 3) * 6 * 512 + (size_t)lane * 16;
      __builtin_prefetch(wtap + 3 * 6 * 512, 0, 1);  // next tap's A block
#pragma unroll
      for (int icb = 0; icb < 3; ++icb) {
        v16h bA = *reinterpret_cast<const v16h*>(rowp + cA + icb * 32);
        v16h bB = *reinterpret_cast<const v16h*>(rowp + cB + icb * 32);
        if (!vA) bA = vzero;  // per-lane select, EXEC restored before WMMA
        if (!vB) bB = vzero;
        const _Float16* wp = wtap + ((size_t)icb * 6 + ocset * 3) * 512;
        v16h a0 = *reinterpret_cast<const v16h*>(wp);
        v16h a1 = *reinterpret_cast<const v16h*>(wp + 512);
        v16h a2 = *reinterpret_cast<const v16h*>(wp + 1024);
        acc[0] = __builtin_amdgcn_wmma_f32_16x16x32_f16(false, a0, false, bA,
                                                        (short)0, acc[0], false, false);
        acc[1] = __builtin_amdgcn_wmma_f32_16x16x32_f16(false, a1, false, bA,
                                                        (short)0, acc[1], false, false);
        acc[2] = __builtin_amdgcn_wmma_f32_16x16x32_f16(false, a2, false, bA,
                                                        (short)0, acc[2], false, false);
        acc[3] = __builtin_amdgcn_wmma_f32_16x16x32_f16(false, a0, false, bB,
                                                        (short)0, acc[3], false, false);
        acc[4] = __builtin_amdgcn_wmma_f32_16x16x32_f16(false, a1, false, bB,
                                                        (short)0, acc[4], false, false);
        acc[5] = __builtin_amdgcn_wmma_f32_16x16x32_f16(false, a2, false, bB,
                                                        (short)0, acc[5], false, false);
      }
    }
  }

  // Epilogue: C layout -> lane holds M = r + rofs for r=0..7 at fixed N.
  // 8 consecutive oc == 8 contiguous halves in NHWC -> b128 loads/stores.
  const int rofs = (lane >> 4) * 8;
#pragma unroll
  for (int g = 0; g < 2; ++g) {
    const int px = x0 + pg2 * 32 + g * 16 + nl;
    const size_t pbase = img + ((size_t)y * IMG + px) * CH;
#pragma unroll
    for (int i = 0; i < 3; ++i) {
      const int ocb = (ocset * 3 + i) * 16 + rofs;
      v8f a = acc[g * 3 + i];
      v8h sv = {};
      if (skip) sv = *reinterpret_cast<const v8h*>(skip + pbase + ocb);
      v8h r;
#pragma unroll
      for (int rr = 0; rr < 8; ++rr) {
        float v = a[rr] + bias[ocb + rr];
        if (skip) v += (float)sv[rr];
        r[rr] = (_Float16)fmaxf(v, 0.f);
      }
      *reinterpret_cast<v8h*>(out + pbase + ocb) = r;
    }
  }
}

// ---------------------------------------------------------------------------
// RED-CNN tail: 96 -> 1, 5x5, + r1 residual, ReLU, f32 output. NHWC input.
// ---------------------------------------------------------------------------
__global__ __launch_bounds__(256) void conv_tail(const _Float16* __restrict__ act,
                                                 const float* __restrict__ wN,
                                                 const float* __restrict__ bN,
                                                 const float* __restrict__ r1,
                                                 float* __restrict__ out) {
  __shared__ float ws[CH * 25];
  int t = threadIdx.x;
  for (int i = t; i < CH * 25; i += 256) ws[i] = wN[i];
  __syncthreads();
  int pix = blockIdx.x * 256 + t;
  int b = blockIdx.y;
  int y = pix >> 8, x = pix & 255;
  const _Float16* ab = act + (size_t)b * CH * PIX;
  float acc = bN[0] + r1[(size_t)b * PIX + pix];
#pragma unroll
  for (int ky = 0; ky < 5; ++ky) {
    int gy = y + ky - 2;
    if (gy < 0 || gy > IMG - 1) continue;
#pragma unroll
    for (int kx = 0; kx < 5; ++kx) {
      int gx = x + kx - 2;
      if (gx < 0 || gx > IMG - 1) continue;
      const _Float16* ap = ab + ((size_t)gy * IMG + gx) * CH;
      int k = ky * 5 + kx;
      for (int ic = 0; ic < CH; ++ic)
        acc = fmaf((float)ap[ic], ws[ic * 25 + k], acc);
    }
  }
  out[(size_t)b * PIX + pix] = fmaxf(acc, 0.f);
}

// ---------------------------------------------------------------------------
extern "C" void kernel_launch(void* const* d_in, const int* in_sizes, int n_in,
                              void* d_out, int out_size, void* d_ws, size_t ws_size,
                              hipStream_t stream) {
  (void)in_sizes; (void)n_in; (void)out_size; (void)ws_size;
  const float* p      = (const float*)d_in[0];
  const float* projw  = (const float*)d_in[1];
  const float* ramp   = (const float*)d_in[2];
  const float* views  = (const float*)d_in[3];
  const float* conv_w = (const float*)d_in[4];
  const float* conv_b = (const float*)d_in[5];
  const float* enc_w0 = (const float*)d_in[6];
  const float* enc_w  = (const float*)d_in[7];
  const float* enc_b  = (const float*)d_in[8];
  const float* dec_w  = (const float*)d_in[9];
  const float* dec_wN = (const float*)d_in[10];
  const float* dec_b  = (const float*)d_in[11];
  const float* dec_bN = (const float*)d_in[12];
  float* out = (float*)d_out;

  char* ws = (char*)d_ws;
  size_t off = 0;
  auto alloc = [&](size_t bytes) -> void* {
    size_t a = (off + 255) & ~(size_t)255;
    off = a + bytes;
    return (void*)(ws + a);
  };
  float*    filtered = (float*)   alloc((size_t)NB * NV * ND * 4);
  float*    xbp      = (float*)   alloc((size_t)NB * NV * PIX * 4);
  float*    xin      = (float*)   alloc((size_t)NB * PIX * 4);
  _Float16* wpack    = (_Float16*)alloc((size_t)8 * 230400 * 2);
  _Float16* actA     = (_Float16*)alloc((size_t)NB * CH * PIX * 2);
  _Float16* actB     = (_Float16*)alloc((size_t)NB * CH * PIX * 2);
  _Float16* actC     = (_Float16*)alloc((size_t)NB * CH * PIX * 2);
  _Float16* actD     = (_Float16*)alloc((size_t)NB * CH * PIX * 2);

  // Repack 96x96 conv weights into WMMA A-fragment layout (f16).
  for (int l = 0; l < 4; ++l)
    repack_w96<<<900, 256, 0, stream>>>(enc_w + (size_t)l * 230400,
                                        wpack + (size_t)l * 230400);
  for (int l = 0; l < 4; ++l)
    repack_w96<<<900, 256, 0, stream>>>(dec_w + (size_t)l * 230400,
                                        wpack + (size_t)(4 + l) * 230400);

  // FBP front-end.
  sino_filter<<<dim3(NV, NB), 256, 0, stream>>>(p, projw, ramp, filtered);
  backproject<<<dim3(PIX / 256, NV, NB), 256, 0, stream>>>(filtered, views, xbp);
  sort_views<<<dim3(PIX / 32, NB), 256, 0, stream>>>(xbp);
  conv_views<<<dim3(PIX / 256, NB), 256, 0, stream>>>(xbp, conv_w, conv_b, xin);

  // RED-CNN (NHWC f16 activations).
  conv_head<<<dim3(PIX / 256, NB), 256, 0, stream>>>(xin, enc_w0, enc_b, actA);

  auto C96 = [&](const _Float16* src, int widx, const float* bias,
                 const _Float16* skip, _Float16* dst) {
    conv96_wmma<<<dim3(IMG / 128, IMG, NB), 256, 0, stream>>>(
        src, wpack + (size_t)widx * 230400, bias, skip, dst);
  };
  C96(actA, 0, enc_b + 96,  nullptr, actB);  // L2  (r2 = actB, kept live)
  C96(actB, 1, enc_b + 192, nullptr, actC);  // L3
  C96(actC, 2, enc_b + 288, nullptr, actD);  // L4  (r3 = actD, kept live)
  C96(actD, 3, enc_b + 384, nullptr, actA);  // L5
  C96(actA, 4, dec_b + 0,   actD,    actC);  // L6  (+r3)
  C96(actC, 5, dec_b + 96,  nullptr, actA);  // L7
  C96(actA, 6, dec_b + 192, actB,    actC);  // L8  (+r2)
  C96(actC, 7, dec_b + 288, nullptr, actA);  // L9

  conv_tail<<<dim3(PIX / 256, NB), 256, 0, stream>>>(actA, dec_wN, dec_bN, xin, out);
}
